// GAT_En2_29755533426832
// MI455X (gfx1250) — matrix operand order repaired and checked by reading, not compiled, not run on hardware
//
#include <hip/hip_runtime.h>

// ---------------------------------------------------------------------------
// GAT 2-layer forward for gfx1250 (MI455X), wave32 + v_wmma_f32_16x16x32_bf16.
//
// Pipeline (all on `stream`):
//   1. x -> bf16                      (elementwise convert)
//   2. W1,W2 -> W^T bf16              (transpose-convert, for contiguous B loads)
//   3. Wh1 = x @ W1   (WMMA bf16)     -> Wh1 f32 + Wh1^T bf16
//   4. a_src/a_dst = Wh1 . a1         (small dot kernel)
//   5. softmax stats m,s per row      (online max/sum streaming over adj)
//   6. h1 = ELU( softmax(e) @ Wh1 )   (fused attention WMMA GEMM) -> h1 bf16
//   7..10. same for layer 2           -> d_out f32 [4096 x 128]
//
// Workspace use: ~16 MiB of d_ws (bump-allocated below).
// ---------------------------------------------------------------------------

typedef __bf16 bf16_t;
typedef __attribute__((ext_vector_type(16))) __bf16 v16bf;
typedef __attribute__((ext_vector_type(8)))  __bf16 v8bf;
typedef __attribute__((ext_vector_type(8)))  float  v8f;
typedef __attribute__((ext_vector_type(4)))  float  v4f;

#define GAT_N      4096
#define GAT_H      4
#define GAT_F1     512
#define GAT_O1     64
#define GAT_F2     256   // = H * O1
#define GAT_O2     32
#define GAT_NE     128   // = H * O2
#define GAT_ALPHA  0.2f
#define GAT_NEGINF (-9.0e15f)

// ------------------------------- utility kernels ---------------------------

__global__ void cvt_f32_bf16(const float* __restrict__ src,
                             bf16_t* __restrict__ dst, int n) {
  int i = blockIdx.x * blockDim.x + threadIdx.x;
  if (i < n) dst[i] = (bf16_t)src[i];
}

// W [H][K][O] f32  ->  WT [H][O][K] bf16
__global__ void wT_bf16(const float* __restrict__ W, bf16_t* __restrict__ WT,
                        int H, int K, int O) {
  int i = blockIdx.x * blockDim.x + threadIdx.x;
  if (i >= H * K * O) return;
  int o = i % O;
  int k = (i / O) % K;
  int h = i / (O * K);
  WT[((size_t)h * O + o) * K + k] = (bf16_t)W[i];
}

// asrc[h][i] = Wh[h][i][:] . a[h][0:Fo] ; adst = . a[h][Fo:2Fo]
__global__ void adot_kernel(const float* __restrict__ Wh,
                            const float* __restrict__ avec,
                            float* __restrict__ asrc, float* __restrict__ adst,
                            int N, int Fo, int H) {
  int idx = blockIdx.x * blockDim.x + threadIdx.x;
  if (idx >= H * N) return;
  int h = idx / N;
  const float* row = Wh + (size_t)idx * Fo;
  const float* a   = avec + (size_t)h * 2 * Fo;
  float s1 = 0.f, s2 = 0.f;
  for (int o = 0; o < Fo; ++o) {
    float v = row[o];
    s1 += v * a[o];
    s2 += v * a[Fo + o];
  }
  asrc[idx] = s1;
  adst[idx] = s2;
}

// Online softmax stats over one adjacency row, all 4 heads at once.
// m[h][i] = max_j e_hij ; s[h][i] = sum_j exp(e_hij - m)   (NEG_INF entries
// contribute exp() == 0 whenever any neighbor exists, matching the reference).
__global__ void att_stats(const float* __restrict__ adj,
                          const float* __restrict__ asrc,
                          const float* __restrict__ adst,
                          float* __restrict__ mOut, float* __restrict__ sOut,
                          int N) {
  const int i   = blockIdx.x;
  const int tid = threadIdx.x;
  float m[GAT_H], s[GAT_H], as[GAT_H];
#pragma unroll
  for (int h = 0; h < GAT_H; ++h) {
    m[h]  = -3.0e38f;
    s[h]  = 0.f;
    as[h] = asrc[h * N + i];
  }
  const float* arow = adj + (size_t)i * N;
  for (int j = tid; j < N; j += 256) {
    float av  = arow[j];
    bool conn = av > 0.f;
#pragma unroll
    for (int h = 0; h < GAT_H; ++h) {
      float e = as[h] + adst[h * N + j];
      e = e > 0.f ? e : GAT_ALPHA * e;
      e = conn ? e : GAT_NEGINF;
      float mn = fmaxf(m[h], e);
      s[h] = s[h] * __expf(m[h] - mn) + __expf(e - mn);
      m[h] = mn;
    }
  }
  __shared__ float sm[GAT_H][256];
  __shared__ float ss[GAT_H][256];
#pragma unroll
  for (int h = 0; h < GAT_H; ++h) { sm[h][tid] = m[h]; ss[h][tid] = s[h]; }
  __syncthreads();
  for (int w = 128; w > 0; w >>= 1) {
    if (tid < w) {
#pragma unroll
      for (int h = 0; h < GAT_H; ++h) {
        float m1 = sm[h][tid], m2 = sm[h][tid + w];
        float M  = fmaxf(m1, m2);
        ss[h][tid] = ss[h][tid] * __expf(m1 - M) + ss[h][tid + w] * __expf(m2 - M);
        sm[h][tid] = M;
      }
    }
    __syncthreads();
  }
  if (tid < GAT_H) {
    mOut[tid * N + i] = sm[tid][0];
    sOut[tid * N + i] = ss[tid][0];
  }
}

// ------------------------------ WMMA GEMM kernels --------------------------
// One wave per (16-row tile, head); OT output tiles of 16 columns each are
// accumulated in-register so the A fragment is loaded/built once per K step.

// Wh[h] = A[N x K] @ B[h][K x Fo]; BT is B transposed ([h][Fo][K], bf16).
// Writes Wh f32 (row major, for the a-vector dots) and Wh^T bf16 (B operand
// of the attention GEMM).
template <int K, int OT>
__global__ __launch_bounds__(32) void gemm_xw(
    const bf16_t* __restrict__ A, const bf16_t* __restrict__ BT,
    float* __restrict__ Wh, bf16_t* __restrict__ WhT, int N) {
  const int lane   = threadIdx.x;
  const int laneLo = lane & 15;
  const int laneHi = lane >> 4;
  const int i0     = blockIdx.x * 16;
  const int h      = blockIdx.y;
  const int Fo     = OT * 16;

  const bf16_t* Arow = A + (size_t)(i0 + laneLo) * K;
  const bf16_t* BTh  = BT + (size_t)h * Fo * K;

  v8f acc[OT] = {};
  for (int kb = 0; kb < K; kb += 32) {
    // A fragment: lane row = lane%16; K = kb + 16*laneHi + 0..15
    // (contiguous, matching the B fragment).
    v8bf alo = *(const v8bf*)(Arow + kb + 16 * laneHi);
    v8bf ahi = *(const v8bf*)(Arow + kb + 16 * laneHi + 8);
    v16bf a;
#pragma unroll
    for (int e = 0; e < 8; ++e) { a[e] = alo[e]; a[8 + e] = ahi[e]; }
#pragma unroll
    for (int t = 0; t < OT; ++t) {
      // B fragment: column n = lane%16, K = kb + 16*laneHi + 0..15 ->
      // 32 contiguous bytes of BT row n.
      const bf16_t* brow =
          BTh + (size_t)(t * 16 + laneLo) * K + kb + 16 * laneHi;
      v16bf b = *(const v16bf*)brow;
      acc[t] = __builtin_amdgcn_wmma_f32_16x16x32_bf16(
          false, a, false, b, (short)0, acc[t], false, false);
    }
  }
  // C/D layout: VGPR e of lane -> row (e + 8*laneHi), col lane%16.
#pragma unroll
  for (int t = 0; t < OT; ++t) {
#pragma unroll
    for (int e = 0; e < 8; ++e) {
      int mrow = e + 8 * laneHi;
      int o    = t * 16 + laneLo;
      float v  = acc[t][e];
      Wh[((size_t)h * N + i0 + mrow) * Fo + o]        = v;
      WhT[((size_t)h * Fo + o) * N + i0 + mrow]       = (bf16_t)v;
    }
  }
}

__device__ __forceinline__ float att_val(float adjv, float ad, float as,
                                         float m) {
  float e = as + ad;
  e = e > 0.f ? e : GAT_ALPHA * e;
  e = adjv > 0.f ? e : GAT_NEGINF;
  return __expf(e - m);  // masked -> 0 unless the whole row is masked
}

// h[i,:] = ELU( (sum_j exp(e_ij - m_i) * Wh[j,:]) / s_i ), fused: the
// attention A fragment is built in-register from adj / a_src / a_dst.
template <int OT>
__global__ __launch_bounds__(32) void att_gemm(
    const float* __restrict__ adj, const float* __restrict__ asrc,
    const float* __restrict__ adst, const float* __restrict__ mRow,
    const float* __restrict__ sRow, const bf16_t* __restrict__ WhT,
    float* __restrict__ outF, bf16_t* __restrict__ outB, int N, int ostride) {
  const int lane   = threadIdx.x;
  const int laneLo = lane & 15;
  const int laneHi = lane >> 4;
  const int i0     = blockIdx.x * 16;
  const int h      = blockIdx.y;
  const int Fo     = OT * 16;

  const int    iA   = i0 + laneLo;  // this lane's attention row (A operand)
  const float  as_i = asrc[h * N + iA];
  const float  m_i  = mRow[h * N + iA];
  const float* arow = adj + (size_t)iA * N;
  const float* adh  = adst + (size_t)h * N;
  const bf16_t* WhTh = WhT + (size_t)h * Fo * N;

  v8f acc[OT] = {};
  for (int jb = 0; jb < N; jb += 32) {
    const int j0 = jb + 16 * laneHi;      // elements 0..7 of the A frag
    const int j1 = jb + 16 * laneHi + 8;  // elements 8..15
    v4f av0 = *(const v4f*)(arow + j0);
    v4f av1 = *(const v4f*)(arow + j0 + 4);
    v4f av2 = *(const v4f*)(arow + j1);
    v4f av3 = *(const v4f*)(arow + j1 + 4);
    v4f ad0 = *(const v4f*)(adh + j0);
    v4f ad1 = *(const v4f*)(adh + j0 + 4);
    v4f ad2 = *(const v4f*)(adh + j1);
    v4f ad3 = *(const v4f*)(adh + j1 + 4);
    v16bf a;
#pragma unroll
    for (int e = 0; e < 4; ++e) {
      a[e]      = (bf16_t)att_val(av0[e], ad0[e], as_i, m_i);
      a[4 + e]  = (bf16_t)att_val(av1[e], ad1[e], as_i, m_i);
      a[8 + e]  = (bf16_t)att_val(av2[e], ad2[e], as_i, m_i);
      a[12 + e] = (bf16_t)att_val(av3[e], ad3[e], as_i, m_i);
    }
#pragma unroll
    for (int t = 0; t < OT; ++t) {
      const bf16_t* brow =
          WhTh + (size_t)(t * 16 + laneLo) * N + jb + 16 * laneHi;
      v16bf b = *(const v16bf*)brow;
      acc[t] = __builtin_amdgcn_wmma_f32_16x16x32_bf16(
          false, a, false, b, (short)0, acc[t], false, false);
    }
  }

  float invs[8];
#pragma unroll
  for (int e = 0; e < 8; ++e)
    invs[e] = 1.0f / sRow[(size_t)h * N + i0 + e + 8 * laneHi];

#pragma unroll
  for (int t = 0; t < OT; ++t) {
#pragma unroll
    for (int e = 0; e < 8; ++e) {
      int   i   = i0 + e + 8 * laneHi;
      float v   = acc[t][e] * invs[e];
      v         = v > 0.f ? v : (__expf(v) - 1.0f);  // ELU
      int   col = h * Fo + t * 16 + laneLo;
      if (outF) outF[(size_t)i * ostride + col] = v;
      if (outB) outB[(size_t)i * ostride + col] = (bf16_t)v;
    }
  }
}

// --------------------------------- driver ----------------------------------

extern "C" void kernel_launch(void* const* d_in, const int* in_sizes, int n_in,
                              void* d_out, int out_size, void* d_ws,
                              size_t ws_size, hipStream_t stream) {
  const float* x   = (const float*)d_in[0];  // [4096,512]
  const float* adj = (const float*)d_in[1];  // [4096,4096]
  const float* W1  = (const float*)d_in[2];  // [4,512,64]
  const float* a1  = (const float*)d_in[3];  // [4,128]
  const float* W2  = (const float*)d_in[4];  // [4,256,32]
  const float* a2  = (const float*)d_in[5];  // [4,64]
  float* out = (float*)d_out;                // [4096,128]

  const int N = GAT_N, H = GAT_H;

  // bump allocator over d_ws (256B aligned), ~16 MiB total
  char* wp = (char*)d_ws;
  auto alloc = [&](size_t bytes) {
    char* p = wp;
    wp += (bytes + 255) & ~(size_t)255;
    return p;
  };
  bf16_t* xb    = (bf16_t*)alloc(sizeof(bf16_t) * N * GAT_F1);
  bf16_t* h1b   = (bf16_t*)alloc(sizeof(bf16_t) * N * GAT_F2);
  bf16_t* W1T   = (bf16_t*)alloc(sizeof(bf16_t) * H * GAT_O1 * GAT_F1);
  bf16_t* W2T   = (bf16_t*)alloc(sizeof(bf16_t) * H * GAT_O2 * GAT_F2);
  bf16_t* WhT1  = (bf16_t*)alloc(sizeof(bf16_t) * H * GAT_O1 * N);
  bf16_t* WhT2  = (bf16_t*)alloc(sizeof(bf16_t) * H * GAT_O2 * N);
  float*  Wh1   = (float*)alloc(sizeof(float) * H * N * GAT_O1);
  float*  Wh2   = (float*)alloc(sizeof(float) * H * N * GAT_O2);
  float*  asrc1 = (float*)alloc(sizeof(float) * H * N);
  float*  adst1 = (float*)alloc(sizeof(float) * H * N);
  float*  m1    = (float*)alloc(sizeof(float) * H * N);
  float*  s1    = (float*)alloc(sizeof(float) * H * N);
  float*  asrc2 = (float*)alloc(sizeof(float) * H * N);
  float*  adst2 = (float*)alloc(sizeof(float) * H * N);
  float*  m2    = (float*)alloc(sizeof(float) * H * N);
  float*  s2    = (float*)alloc(sizeof(float) * H * N);

  // 1) convert inputs
  {
    int n = N * GAT_F1;
    cvt_f32_bf16<<<(n + 255) / 256, 256, 0, stream>>>(x, xb, n);
  }
  {
    int n = H * GAT_F1 * GAT_O1;
    wT_bf16<<<(n + 255) / 256, 256, 0, stream>>>(W1, W1T, H, GAT_F1, GAT_O1);
  }
  {
    int n = H * GAT_F2 * GAT_O2;
    wT_bf16<<<(n + 255) / 256, 256, 0, stream>>>(W2, W2T, H, GAT_F2, GAT_O2);
  }

  // ---------- layer 1 ----------
  gemm_xw<GAT_F1, 4><<<dim3(N / 16, H), 32, 0, stream>>>(xb, W1T, Wh1, WhT1, N);
  adot_kernel<<<(H * N + 255) / 256, 256, 0, stream>>>(Wh1, a1, asrc1, adst1, N,
                                                       GAT_O1, H);
  att_stats<<<N, 256, 0, stream>>>(adj, asrc1, adst1, m1, s1, N);
  att_gemm<4><<<dim3(N / 16, H), 32, 0, stream>>>(adj, asrc1, adst1, m1, s1,
                                                  WhT1, nullptr, h1b, N,
                                                  GAT_F2);

  // ---------- layer 2 ----------
  gemm_xw<GAT_F2, 2><<<dim3(N / 16, H), 32, 0, stream>>>(h1b, W2T, Wh2, WhT2,
                                                         N);
  adot_kernel<<<(H * N + 255) / 256, 256, 0, stream>>>(Wh2, a2, asrc2, adst2, N,
                                                       GAT_O2, H);
  att_stats<<<N, 256, 0, stream>>>(adj, asrc2, adst2, m2, s2, N);
  att_gemm<2><<<dim3(N / 16, H), 32, 0, stream>>>(adj, asrc2, adst2, m2, s2,
                                                  WhT2, out, nullptr, N,
                                                  GAT_NE);
}
